// QuantMLP_1984274891104
// MI455X (gfx1250) — matrix operand order repaired and checked by reading, not compiled, not run on hardware
//
#include <hip/hip_runtime.h>
#include <hip/hip_bf16.h>

// ---------------------------------------------------------------------------
// QuantMLP for MI455X (gfx1250): 4x [quant-e4m3 -> bf16 WMMA GEMM -> bias ->
// GELU -> quant] with v_wmma_f32_16x16x32_bf16 and async global->LDS staging.
// Block tile 128x256 (8 waves), wave tile 64x64 (4x4 WMMA tiles).
// K-loop kept rolled (unroll 1) so the 16 f32 accumulators stay pinned
// in-place across iterations (no v_mov register rotation, <256 VGPRs).
// ---------------------------------------------------------------------------

#define TDIM 8192
#define DDIM 4096
#define BM 128
#define BN 256
#define BK 32
#define LDL 40              // LDS row stride in bf16 elems (BK + 8 pad, bank-conflict free)
#define KT (DDIM / BK)      // 128 k-steps

#if defined(__HIP_DEVICE_COMPILE__) && __has_builtin(__builtin_amdgcn_global_load_async_to_lds_b128)
#define HAVE_ASYNC 1
#else
#define HAVE_ASYNC 0
#endif

typedef __attribute__((ext_vector_type(16))) __bf16 v16bf;
typedef __attribute__((ext_vector_type(8)))  float  v8f;
typedef __attribute__((ext_vector_type(4)))  int    v4i;

union FragAB { v16bf v; uint4 q[2]; };

// ---- fp8 e4m3fn round-to-nearest-even quantization (value-preserving in bf16)
__device__ __forceinline__ float quant_e4m3(float x) {
  float ax = fabsf(x);
  if (ax >= 448.0f) return copysignf(448.0f, x);       // saturate
  unsigned eb = (__float_as_uint(ax) >> 23) & 0xFFu;   // biased exponent
  if (eb < 121u) {                                     // |x| < 2^-6 : subnormal grid 2^-9
    return rintf(x * 512.0f) * 0.001953125f;
  }
  float istep = __uint_as_float((257u - eb) << 23);    // 2^-(e-3)
  float step  = __uint_as_float((eb - 3u) << 23);      // 2^(e-3) grid spacing
  float q = rintf(x * istep) * step;
  return fminf(fmaxf(q, -448.0f), 448.0f);
}

__device__ __forceinline__ float gelu_exact(float x) {
  return 0.5f * x * (1.0f + erff(x * 0.70710678118654752440f));
}

__device__ __forceinline__ void wait_async0() {
#if __has_builtin(__builtin_amdgcn_s_wait_asynccnt)
  __builtin_amdgcn_s_wait_asynccnt(0);
#else
  asm volatile("s_wait_asynccnt 0" ::: "memory");
#endif
}

#if HAVE_ASYNC
// Async 16B global->LDS copy. Builtin signature (probe-verified):
//   void(v4i addrspace(1)*, v4i addrspace(3)*, imm int offset, imm int cpol)
__device__ __forceinline__ void async_b128(const void* g, void* l) {
  __builtin_amdgcn_global_load_async_to_lds_b128(
      (__attribute__((address_space(1))) v4i*)g,
      (__attribute__((address_space(3))) v4i*)l, 0, 0);
}
#endif

// Stage one ROWSx32 bf16 tile (g row stride DDIM elems) into LDS (row stride LDL).
// 256 threads x NPASS passes x 16B.
template <int NPASS>
__device__ __forceinline__ void stage_tile(const __hip_bfloat16* __restrict__ g,
                                           __bf16* __restrict__ l, int tid) {
#pragma unroll
  for (int p = 0; p < NPASS; ++p) {
    int flat = tid + p * 256;
    int row  = flat >> 2;          // 4 x 16B chunks per 64B row
    int col  = (flat & 3) * 8;     // elem offset
#if HAVE_ASYNC
    async_b128(g + (size_t)row * DDIM + col, l + row * LDL + col);
#else
    *(uint4*)(l + row * LDL + col) = *(const uint4*)(g + (size_t)row * DDIM + col);
#endif
  }
}

// ---------------------------------------------------------------------------
// Elementwise pre-quantization of the first layer input: out = e4m3(x / s0).
// Vectorized: 8 bf16 per thread via uint4.
// ---------------------------------------------------------------------------
__global__ void quant_in_kernel(const __hip_bfloat16* __restrict__ x,
                                __hip_bfloat16* __restrict__ o,
                                const float* __restrict__ scales, int n8) {
  int i = blockIdx.x * blockDim.x + threadIdx.x;
  if (i < n8) {
    const float inv = 1.0f / scales[0];
    uint4 v = *(const uint4*)(x + (size_t)i * 8);
    __hip_bfloat16 e[8];
    *(uint4*)e = v;
#pragma unroll
    for (int j = 0; j < 8; ++j)
      e[j] = __float2bfloat16(quant_e4m3(__bfloat162float(e[j]) * inv));
    *(uint4*)(o + (size_t)i * 8) = *(uint4*)e;
  }
}

// ---------------------------------------------------------------------------
// One layer: out[t][o] = act( f32(sum_d Aq[t][d] * W[o][d]) * s_l + b[o] )
// ---------------------------------------------------------------------------
__global__ __launch_bounds__(256, 1)
void qgemm_layer_kernel(const __hip_bfloat16* __restrict__ Aq,
                        const __hip_bfloat16* __restrict__ Wl,
                        const __hip_bfloat16* __restrict__ bias,
                        const float* __restrict__ scales,
                        int layer, int apply_act,
                        __hip_bfloat16* __restrict__ out) {
  __shared__ __align__(16) __bf16 As[2][BM * LDL];   // 2 x 10KB
  __shared__ __align__(16) __bf16 Bs[2][BN * LDL];   // 2 x 20KB  (total 60KB)

  const int tid  = threadIdx.x;
  const int lane = tid & 31;
  const int wav  = tid >> 5;
  const int wm   = wav >> 2;       // 0..1 : 64-row slab
  const int wn   = wav & 3;        // 0..3 : 64-col slab
  const int mrow = lane & 15;      // M (or N for B) row within 16x16 tile
  const int kh   = lane >> 4;      // K half selector per WMMA VGPR layout

  const int m0 = blockIdx.y * BM;
  const int n0 = blockIdx.x * BN;

  const __hip_bfloat16* Ag = Aq + (size_t)m0 * DDIM;   // rows t, contiguous in d
  const __hip_bfloat16* Wg = Wl + (size_t)n0 * DDIM;   // rows o, contiguous in d

  // Per-lane LDS fragment base offsets (in elems), before buffer selection.
  const int aoffA = (wm * 64 + mrow) * LDL + kh * 8;        // + tm*16*LDL
  const int aoffB = (wn * 64 + mrow) * LDL + kh * 8;        // + tn*16*LDL

  const v8f vzero = {0.f, 0.f, 0.f, 0.f, 0.f, 0.f, 0.f, 0.f};
  v8f acc[4][4];
#pragma unroll
  for (int tm = 0; tm < 4; ++tm)
#pragma unroll
    for (int tn = 0; tn < 4; ++tn) acc[tm][tn] = vzero;

#if HAVE_ASYNC
  // ---- prologue: fill buffer 0
  stage_tile<2>(Ag, &As[0][0], tid);
  stage_tile<4>(Wg, &Bs[0][0], tid);

#pragma unroll 1   // keep rolled: accumulators stay pinned in-place
  for (int kt = 0; kt < KT; ++kt) {
    const int cur = kt & 1;
    wait_async0();            // async writes into buf `cur` have landed in LDS
    __syncthreads();          // everyone done reading buf cur^1 from last iter

    if (kt + 1 < KT) {        // prefetch next K-slab into the other buffer
      stage_tile<2>(Ag + (size_t)(kt + 1) * BK, &As[cur ^ 1][0], tid);
      stage_tile<4>(Wg + (size_t)(kt + 1) * BK, &Bs[cur ^ 1][0], tid);
    }
#else
#pragma unroll 1
  for (int kt = 0; kt < KT; ++kt) {
    const int cur = 0;
    __syncthreads();
    stage_tile<2>(Ag + (size_t)kt * BK, &As[0][0], tid);
    stage_tile<4>(Wg + (size_t)kt * BK, &Bs[0][0], tid);
    __syncthreads();
#endif

    // ---- fragment loads from LDS (conflict-free b128 reads) ----
    const __bf16* abase = &As[cur][aoffA];
    const __bf16* bbase = &Bs[cur][aoffB];

    FragAB afr[4], bfr[4];
#pragma unroll
    for (int tm = 0; tm < 4; ++tm) {
      // A 16-bit layout: lanes 0-15 hold K 0..7 & 16..23 ; lanes 16-31 hold K 8..15 & 24..31
      afr[tm].q[0] = *(const uint4*)(abase + tm * (16 * LDL));
      afr[tm].q[1] = *(const uint4*)(abase + tm * (16 * LDL) + 16);
    }
#pragma unroll
    for (int tn = 0; tn < 4; ++tn) {
      // B layout: lane holds column n = lane&15; lanes 0-15 K 0..7 & 16..23, lanes 16-31 K 8..15 & 24..31
      bfr[tn].q[0] = *(const uint4*)(bbase + tn * (16 * LDL));
      bfr[tn].q[1] = *(const uint4*)(bbase + tn * (16 * LDL) + 16);
    }

    // ---- 16 WMMAs per wave per k-step ----
#pragma unroll
    for (int tm = 0; tm < 4; ++tm)
#pragma unroll
      for (int tn = 0; tn < 4; ++tn)
        acc[tm][tn] = __builtin_amdgcn_wmma_f32_16x16x32_bf16(
            false, afr[tm].v, false, bfr[tn].v, (short)0, acc[tm][tn],
            false, false);
  }

  // ---- epilogue: scale + bias -> bf16 -> [gelu -> quant(/s_next)] -> store ----
  const float scale    = scales[layer];
  const float inv_next = apply_act ? (1.0f / scales[layer + 1]) : 1.0f;

#pragma unroll
  for (int tm = 0; tm < 4; ++tm) {
#pragma unroll
    for (int tn = 0; tn < 4; ++tn) {
      const int gn = n0 + wn * 64 + tn * 16 + mrow;
      const float bv = __bfloat162float(bias[gn]);
#pragma unroll
      for (int j = 0; j < 8; ++j) {
        // C/D layout: VGPR j -> M = j (lanes 0-15) or 8+j (lanes 16-31); N = lane&15
        const int gm = m0 + wm * 64 + tm * 16 + kh * 8 + j;
        float v = acc[tm][tn][j] * scale + bv;
        __hip_bfloat16 hb = __float2bfloat16(v);
        if (apply_act) {
          float g = gelu_exact(__bfloat162float(hb));
          hb = __float2bfloat16(quant_e4m3(g * inv_next));
        }
        out[(size_t)gm * DDIM + gn] = hb;
      }
    }
  }
}

// ---------------------------------------------------------------------------
// Host launch: quantize x, then 4 GEMM layers ping-ponging ws <-> d_out,
// final result copied back into d_out. Needs ws >= 64MB.
// ---------------------------------------------------------------------------
extern "C" void kernel_launch(void* const* d_in, const int* in_sizes, int n_in,
                              void* d_out, int out_size, void* d_ws, size_t ws_size,
                              hipStream_t stream) {
  (void)in_sizes; (void)n_in; (void)out_size; (void)ws_size;

  const __hip_bfloat16* x      = (const __hip_bfloat16*)d_in[0];
  const __hip_bfloat16* W      = (const __hip_bfloat16*)d_in[1];
  const __hip_bfloat16* b      = (const __hip_bfloat16*)d_in[2];
  const float*          scales = (const float*)d_in[3];

  __hip_bfloat16* ws  = (__hip_bfloat16*)d_ws;
  __hip_bfloat16* out = (__hip_bfloat16*)d_out;

  const size_t TD = (size_t)TDIM * DDIM;           // 33,554,432
  const size_t WW = (size_t)DDIM * DDIM;           // weights per layer

  const int n8 = (int)(TD / 8);
  quant_in_kernel<<<(n8 + 255) / 256, 256, 0, stream>>>(x, ws, scales, n8);

  dim3 grid(DDIM / BN, TDIM / BM);                 // (16, 64)
  // L0: ws -> out   (gelu + quant for next layer)
  qgemm_layer_kernel<<<grid, 256, 0, stream>>>(ws,  W + 0 * WW, b + 0 * DDIM, scales, 0, 1, out);
  // L1: out -> ws
  qgemm_layer_kernel<<<grid, 256, 0, stream>>>(out, W + 1 * WW, b + 1 * DDIM, scales, 1, 1, ws);
  // L2: ws -> out
  qgemm_layer_kernel<<<grid, 256, 0, stream>>>(ws,  W + 2 * WW, b + 2 * DDIM, scales, 2, 1, out);
  // L3: out -> ws   (no activation, no quant)
  qgemm_layer_kernel<<<grid, 256, 0, stream>>>(out, W + 3 * WW, b + 3 * DDIM, scales, 3, 0, ws);

  (void)hipMemcpyAsync(d_out, ws, TD * sizeof(__hip_bfloat16),
                       hipMemcpyDeviceToDevice, stream);
}
